// Ece_57501022159076
// MI455X (gfx1250) — compile-verified, matrix operand order checked by reading, not run on hardware
//
#include <hip/hip_runtime.h>

// ---------------- problem constants ----------------
#define NCLS   256      // classes
#define NBINS  15       // ECE bins
#define NEDGE  16       // bin edges
#define NTGT   30       // order-statistic targets: rank0, {floor(kN/15), +1} k=1..14, rank N-1
#define H1BINS 2048     // level-1 histogram: key >> 20   (bits 30:20)
#define H2BINS 1024     // level-2 histogram: (key>>10)&1023
#define H3BINS 1024     // level-3 histogram: key & 1023
#define TILE_R 32       // softmax tile rows per block
#define TPAD   260      // padded LDS row stride (1040B: 16B-aligned rows, 2-way worst bank case)
#define CHUNK  16384    // elements per block in streaming per-class passes

typedef float    v2f  __attribute__((ext_vector_type(2)));
typedef float    v8f  __attribute__((ext_vector_type(8)));
typedef _Float16 v16h __attribute__((ext_vector_type(16)));
typedef unsigned uint32x4_t __attribute__((ext_vector_type(4)));
typedef int      int32x4_t  __attribute__((ext_vector_type(4)));
typedef int      int32x8_t  __attribute__((ext_vector_type(8)));

#define AS1 __attribute__((address_space(1)))
#define AS3 __attribute__((address_space(3)))

// CDNA5 async/TDM availability (probe-verified builtins; arity of TDM differs per toolchain)
#if defined(__has_builtin)
# if __has_builtin(__builtin_amdgcn_tensor_load_to_lds) && __has_builtin(__builtin_amdgcn_s_wait_tensorcnt)
#  define USE_TDM 1
# endif
# if __has_builtin(__builtin_amdgcn_global_load_async_to_lds_b128) && __has_builtin(__builtin_amdgcn_s_wait_asynccnt)
#  define USE_ASYNC_LD 1
# endif
# if __has_builtin(__builtin_amdgcn_global_store_async_from_lds_b32)
#  define USE_ASYNC_ST 1
# endif
#endif
#if defined(__has_include)
# if __has_include(<hip/amd_detail/amd_gfx1250_TDM.h>)
#  define TDM_6ARG 1   // therock-10.0 headers -> 6-arg tensor builtin
# endif
#endif

// async builtins take typed pointers: addrspace(1) int* / addrspace(3) int*
__device__ __forceinline__ AS1 int* as_g(const void* p) {
  return (AS1 int*)(unsigned long long)p;
}
__device__ __forceinline__ AS3 int* as_l(const void* p) {
  return (AS3 int*)(unsigned)(unsigned long long)p;
}

// rank of target t (ascending): 0, f(1),f(1)+1, ..., f(14),f(14)+1, N-1
__device__ __forceinline__ unsigned rank_of(int t, unsigned n) {
  if (t == 0) return 0u;
  if (t == NTGT - 1) return n - 1u;
  unsigned k = (unsigned)((t + 1) >> 1);
  unsigned base = (unsigned)(((unsigned long long)k * n) / 15ull);
  return base + ((t & 1) ? 0u : 1u);
}

// ---------------- K0: zero scratch + output ----------------
__global__ void zero_ws(uint4* __restrict__ p, size_t nvec,
                        float* __restrict__ out, int outN) {
  size_t i = (size_t)blockIdx.x * blockDim.x + threadIdx.x;
  if (i < (size_t)outN) out[i] = 0.0f;
  const size_t stride = (size_t)gridDim.x * blockDim.x;
  const uint4 z = make_uint4(0u, 0u, 0u, 0u);
  for (; i < nvec; i += stride) p[i] = z;
}

// ---------------- K1: softmax + transpose (TDM load, WMMA row sums, async store) ----------------
__global__ __launch_bounds__(256) void softmax_transpose(
    const float* __restrict__ logits, float* __restrict__ probsT, int n) {
  __shared__ float tile[TILE_R][TPAD];   // 1040B row stride: 16B aligned rows
  __shared__ float rowMaxS[TILE_R];
  __shared__ float rowSumS[TILE_R];
  __shared__ float rowInvS[TILE_R];
  const int tid  = threadIdx.x;
  const int lane = tid & 31;
  const int wave = tid >> 5;
  const int row0 = blockIdx.x * TILE_R;

  // ---- tile load: TDM DMA > per-lane async-to-LDS > sync loads ----
#if USE_TDM
  if (wave == 0) {
    const unsigned long long ga = (unsigned long long)(logits + (size_t)row0 * NCLS);
    const unsigned ldsa = (unsigned)(unsigned long long)&tile[0][0];
    uint32x4_t g0;
    g0[0] = 1u;                                   // count=1 valid user descriptor
    g0[1] = ldsa;                                 // lds_addr (bytes)
    g0[2] = (unsigned)ga;                         // global_addr[31:0] (tile start)
    g0[3] = (unsigned)((ga >> 32) & 0x1FFFFFFu)   // global_addr[56:32]
          | (2u << 30);                           // type=2 ("image")
    int32x8_t g1;
    g1[0] = (int)((2u << 16)      // data_size = 4B
                | (1u << 20)      // pad_enable: pad LDS rows
                | (7u << 22)      // pad_interval: every 256 DWORDs (one 256-col row)
                | (3u << 25));    // pad_amount: 4 DWORDs -> LDS stride 260 floats
    g1[1] = (int)(256u << 16);    // atomic_barrier addr=0 ; tensor_dim0 lo16 = 256
    g1[2] = (int)(((unsigned)TILE_R) << 16); // tensor_dim0 hi=0 ; tensor_dim1 lo16 = 32
    g1[3] = (int)(256u << 16);    // tensor_dim1 hi=0 ; tile_dim0 = 256
    g1[4] = (int)TILE_R;          // tile_dim1 = 32 ; tile_dim2 = 0
    g1[5] = 256;                  // tensor_dim0_stride = 256 (lo32)
    g1[6] = 0;                    // stride hi16 ; tensor_dim1_stride lo16 = 0
    g1[7] = 0;
    int32x4_t gz = {0, 0, 0, 0};
#if TDM_6ARG
    int32x8_t gz8 = {0, 0, 0, 0, 0, 0, 0, 0};
    __builtin_amdgcn_tensor_load_to_lds(g0, g1, gz, gz, gz8, 0);
#else
    __builtin_amdgcn_tensor_load_to_lds(g0, g1, gz, gz, 0);
#endif
    __builtin_amdgcn_s_wait_tensorcnt(0);
  }
  __syncthreads();
#elif USE_ASYNC_LD
  for (int it = 0; it < 8; ++it) {
    const int p = tid + 256 * it;   // 2048 x 16B chunks
    const int r = p >> 6;
    const int q = p & 63;
    const float* gsrc = logits + (size_t)(row0 + r) * NCLS + q * 4;
    __builtin_amdgcn_global_load_async_to_lds_b128(as_g(gsrc), as_l(&tile[r][q * 4]), 0, 0);
  }
  __builtin_amdgcn_s_wait_asynccnt(0);
  __syncthreads();
#else
  for (int i = 0; i < TILE_R; ++i)
    tile[i][tid] = logits[(size_t)(row0 + i) * NCLS + tid];
  __syncthreads();
#endif

  // row max: wave w owns rows 4w..4w+3, lane-strided + xor-shuffle reduce (wave32)
  for (int j = 0; j < 4; ++j) {
    const int r = wave * 4 + j;
    float m = -3.4e38f;
    for (int q = 0; q < 8; ++q) m = fmaxf(m, tile[r][lane + 32 * q]);
    for (int off = 16; off; off >>= 1) m = fmaxf(m, __shfl_xor(m, off, 32));
    if (lane == 0) rowMaxS[r] = m;
  }
  __syncthreads();

  // exp in place (thread = column)
  for (int i = 0; i < TILE_R; ++i)
    tile[i][tid] = __expf(tile[i][tid] - rowMaxS[i]);
  __syncthreads();

  // row sums on the matrix pipe: D = A(16xK) x ones(Kx16), accumulated over K=256.
  // Waves 0,1 only (uniform branch -> EXEC all ones inside, as WMMA requires).
  if (wave < 2) {
    const int rb  = wave * 16;
    const int m16 = lane & 15;
    v8f acc = {};
#if __has_builtin(__builtin_amdgcn_wmma_f32_16x16x4_f32)
    v2f bones; bones[0] = 1.0f; bones[1] = 1.0f;
    const int kadd = (lane >> 4) * 2;   // lanes 0-15: K=k0,k0+1 ; lanes 16-31: K=k0+2,k0+3
    for (int k0 = 0; k0 < NCLS; k0 += 4) {
      v2f a;
      a[0] = tile[rb + m16][k0 + kadd];
      a[1] = tile[rb + m16][k0 + kadd + 1];
      acc = __builtin_amdgcn_wmma_f32_16x16x4_f32(false, a, false, bones,
                                                  (short)0, acc, false, false);
    }
#else
    v16h bones;
    for (int i = 0; i < 16; ++i) bones[i] = (_Float16)1.0f;
    const int kadd = (lane >> 4) * 16;
    for (int k0 = 0; k0 < NCLS; k0 += 32) {
      v16h a;
      for (int i = 0; i < 16; ++i) a[i] = (_Float16)tile[rb + m16][k0 + kadd + i];
      acc = __builtin_amdgcn_wmma_f32_16x16x32_f16(false, a, false, bones,
                                                   (short)0, acc, false, false);
    }
#endif
    if (lane == 0)  { for (int v = 0; v < 8; ++v) rowSumS[rb + v]     = acc[v]; }
    if (lane == 16) { for (int v = 0; v < 8; ++v) rowSumS[rb + 8 + v] = acc[v]; }
  }
  __syncthreads();
  if (tid < TILE_R) rowInvS[tid] = 1.0f / rowSumS[tid];
  __syncthreads();

  // scale in LDS (thread = column) so the transposed write is a pure LDS->mem copy
  for (int i = 0; i < TILE_R; ++i)
    tile[i][tid] *= rowInvS[i];
  __syncthreads();

  // transposed, coalesced store: wave w writes classes [32w,32w+32), lane = row
  const int c0 = wave * 32;
  for (int cc = 0; cc < 32; ++cc) {
    const int c = c0 + cc;
#if USE_ASYNC_ST
    __builtin_amdgcn_global_store_async_from_lds_b32(
        as_g(&probsT[(size_t)c * n + row0 + lane]),
        as_l(&tile[lane][c]), 0, 0);      // s_endpgm waits all counters
#else
    probsT[(size_t)c * n + row0 + lane] = tile[lane][c];
#endif
  }
}

// ---------------- K2: level-1 histogram (bits 30:20) ----------------
__global__ __launch_bounds__(256) void hist_l1(
    const float* __restrict__ probsT, unsigned* __restrict__ hist1, int n) {
  __shared__ unsigned lh[H1BINS];
  const int tid = threadIdx.x;
  const int c   = blockIdx.y;
  for (int i = tid; i < H1BINS; i += 256) lh[i] = 0;
  __syncthreads();
  const size_t base = (size_t)c * n + (size_t)blockIdx.x * CHUNK;
  const float4* p4 = (const float4*)(probsT + base);
  for (int i = 0; i < CHUNK / 1024; ++i) {
    __builtin_prefetch(p4 + tid + 256 * (i + 2), 0, 1);  // global_prefetch_b8
    const float4 v = p4[tid + 256 * i];
    atomicAdd(&lh[__float_as_uint(v.x) >> 20], 1u);
    atomicAdd(&lh[__float_as_uint(v.y) >> 20], 1u);
    atomicAdd(&lh[__float_as_uint(v.z) >> 20], 1u);
    atomicAdd(&lh[__float_as_uint(v.w) >> 20], 1u);
  }
  __syncthreads();
  unsigned* gh = hist1 + (size_t)c * H1BINS;
  for (int i = tid; i < H1BINS; i += 256)
    if (lh[i]) atomicAdd(&gh[i], lh[i]);
}

// ---------------- K3: per-class sweep -> bucket + residual rank per target ----------------
__global__ void select_l1(const unsigned* __restrict__ hist1,
                          unsigned* __restrict__ tb1, unsigned* __restrict__ tr1, int n) {
  const int c = blockIdx.x;
  if (threadIdx.x != 0) return;
  const unsigned* h = hist1 + (size_t)c * H1BINS;
  unsigned cum = 0; int t = 0;
  for (int b = 0; b < H1BINS && t < NTGT; ++b) {
    const unsigned c2 = cum + h[b];
    while (t < NTGT && rank_of(t, (unsigned)n) < c2) {
      tb1[c * NTGT + t] = (unsigned)b;
      tr1[c * NTGT + t] = rank_of(t, (unsigned)n) - cum;
      ++t;
    }
    cum = c2;
  }
}

// ---------------- K4: level-2 histogram per (class,target) ----------------
__global__ __launch_bounds__(256) void hist_l2(
    const float* __restrict__ probsT, const unsigned* __restrict__ tb1,
    unsigned* __restrict__ hist2, int n) {
  __shared__ unsigned char lutF[H1BINS], lutC[H1BINS];
  __shared__ unsigned tb[NTGT];
  const int tid = threadIdx.x;
  const int c   = blockIdx.y;
  for (int i = tid; i < H1BINS; i += 256) { lutF[i] = 0xFF; lutC[i] = 0; }
  if (tid < NTGT) tb[tid] = tb1[c * NTGT + tid];
  __syncthreads();
  if (tid == 0) { // targets ascending -> equal buckets contiguous
    for (int t = 0; t < NTGT; ++t) {
      const unsigned b = tb[t];
      if (lutF[b] == 0xFF) lutF[b] = (unsigned char)t;
      lutC[b]++;
    }
  }
  __syncthreads();
  const size_t base = (size_t)c * n + (size_t)blockIdx.x * CHUNK;
  unsigned* h2c = hist2 + (size_t)c * NTGT * H2BINS;
  auto h2one = [&](float cf) {
    const unsigned key = __float_as_uint(cf);
    const unsigned b = key >> 20;
    const unsigned f = lutF[b];
    if (f != 0xFF) {
      const unsigned sub = (key >> 10) & (H2BINS - 1);
      const unsigned cnt = lutC[b];
      for (unsigned j = 0; j < cnt; ++j)
        atomicAdd(&h2c[(f + j) * H2BINS + sub], 1u);
    }
  };
  const float4* p4 = (const float4*)(probsT + base);
  for (int i = 0; i < CHUNK / 1024; ++i) {
    const float4 v = p4[tid + 256 * i];
    h2one(v.x); h2one(v.y); h2one(v.z); h2one(v.w);
  }
}

// ---------------- K5: level-2 select ----------------
__global__ void select_l2(const unsigned* __restrict__ hist2,
                          const unsigned* __restrict__ tr1,
                          unsigned* __restrict__ tb2, unsigned* __restrict__ tr2) {
  const int c = blockIdx.x, t = threadIdx.x;
  if (t >= NTGT) return;
  const unsigned* h = hist2 + ((size_t)c * NTGT + t) * H2BINS;
  unsigned r = tr1[c * NTGT + t], cum = 0;
  for (int b = 0; b < H2BINS; ++b) {
    const unsigned hb = h[b];
    if (r < cum + hb) { tb2[c * NTGT + t] = b; tr2[c * NTGT + t] = r - cum; return; }
    cum += hb;
  }
  tb2[c * NTGT + t] = H2BINS - 1; tr2[c * NTGT + t] = 0;
}

// ---------------- K6: level-3 histogram ----------------
__global__ __launch_bounds__(256) void hist_l3(
    const float* __restrict__ probsT, const unsigned* __restrict__ tb1,
    const unsigned* __restrict__ tb2, unsigned* __restrict__ hist3, int n) {
  __shared__ unsigned char lutF[H1BINS], lutC[H1BINS];
  __shared__ unsigned tb[NTGT], tm[NTGT];
  const int tid = threadIdx.x;
  const int c   = blockIdx.y;
  for (int i = tid; i < H1BINS; i += 256) { lutF[i] = 0xFF; lutC[i] = 0; }
  if (tid < NTGT) { tb[tid] = tb1[c * NTGT + tid]; tm[tid] = tb2[c * NTGT + tid]; }
  __syncthreads();
  if (tid == 0) {
    for (int t = 0; t < NTGT; ++t) {
      const unsigned b = tb[t];
      if (lutF[b] == 0xFF) lutF[b] = (unsigned char)t;
      lutC[b]++;
    }
  }
  __syncthreads();
  const size_t base = (size_t)c * n + (size_t)blockIdx.x * CHUNK;
  unsigned* h3c = hist3 + (size_t)c * NTGT * H3BINS;
  auto h3one = [&](float cf) {
    const unsigned key = __float_as_uint(cf);
    const unsigned b = key >> 20;
    const unsigned f = lutF[b];
    if (f != 0xFF) {
      const unsigned mid = (key >> 10) & (H2BINS - 1);
      const unsigned low = key & (H3BINS - 1);
      const unsigned cnt = lutC[b];
      for (unsigned j = 0; j < cnt; ++j) {
        const unsigned t = f + j;
        if (tm[t] == mid) atomicAdd(&h3c[t * H3BINS + low], 1u);
      }
    }
  };
  const float4* p4 = (const float4*)(probsT + base);
  for (int i = 0; i < CHUNK / 1024; ++i) {
    const float4 v = p4[tid + 256 * i];
    h3one(v.x); h3one(v.y); h3one(v.z); h3one(v.w);
  }
}

// ---------------- K7: level-3 select -> exact order-statistic values ----------------
__global__ void select_l3(const unsigned* __restrict__ hist3,
                          const unsigned* __restrict__ tb1, const unsigned* __restrict__ tb2,
                          const unsigned* __restrict__ tr2, float* __restrict__ vals) {
  const int c = blockIdx.x, t = threadIdx.x;
  if (t >= NTGT) return;
  const unsigned* h = hist3 + ((size_t)c * NTGT + t) * H3BINS;
  unsigned r = tr2[c * NTGT + t], cum = 0, b3 = H3BINS - 1;
  for (int b = 0; b < H3BINS; ++b) {
    const unsigned hb = h[b];
    if (r < cum + hb) { b3 = (unsigned)b; break; }
    cum += hb;
  }
  const unsigned key = (tb1[c * NTGT + t] << 20) | (tb2[c * NTGT + t] << 10) | b3;
  vals[c * NTGT + t] = __uint_as_float(key);
}

// ---------------- K8: interp -> 16 edges per class ----------------
__global__ void make_edges(const float* __restrict__ vals,
                           float* __restrict__ edges, int n) {
  const int c = blockIdx.x * blockDim.x + threadIdx.x;
  if (c >= NCLS) return;
  const float* v = vals + c * NTGT;
  float* e = edges + c * NEDGE;
  e[0] = v[0];
  e[NBINS] = v[NTGT - 1];          // interp clamps at xp[-1] -> max
  for (int k = 1; k < NBINS; ++k) {
    const double t  = (double)k * (double)n / 15.0;
    const double fr = t - floor(t);
    const double lo = (double)v[2 * k - 1], hi = (double)v[2 * k];
    e[k] = (float)(lo + fr * (hi - lo));
  }
}

// ---------------- K9: per-bin counts and label counts ----------------
__global__ __launch_bounds__(256) void bin_stats(
    const float* __restrict__ probsT, const int* __restrict__ labels,
    const float* __restrict__ edges, unsigned* __restrict__ binLab,
    unsigned* __restrict__ binCnt, int n) {
  __shared__ unsigned lsum[NEDGE], lcnt[NEDGE];
  const int tid = threadIdx.x;
  const int c   = blockIdx.y;
  if (tid < NEDGE) { lsum[tid] = 0; lcnt[tid] = 0; }
  float e[NEDGE];
  for (int k = 0; k < NEDGE; ++k) e[k] = edges[c * NEDGE + k];
  __syncthreads();
  const size_t base = (size_t)c * n + (size_t)blockIdx.x * CHUNK;
  const float4* p4 = (const float4*)(probsT + base);
  const int4*   l4 = (const int4*)(labels + (size_t)blockIdx.x * CHUNK);
  auto doone = [&](float conf, int lab) {
    int idx = 0;                      // searchsorted(edges, conf, 'left')
#pragma unroll
    for (int k = 0; k < NEDGE; ++k) idx += (e[k] < conf);
    if (idx >= 1) {                   // conf == min -> no bin (matches reference)
      atomicAdd(&lcnt[idx - 1], 1u);
      if (lab == c) atomicAdd(&lsum[idx - 1], 1u);
    }
  };
  for (int i = 0; i < CHUNK / 1024; ++i) {
    __builtin_prefetch(p4 + tid + 256 * (i + 2), 0, 1);
    const int vi = tid + 256 * i;
    const float4 v = p4[vi];
    const int4 lv = l4[vi];
    doone(v.x, lv.x); doone(v.y, lv.y); doone(v.z, lv.z); doone(v.w, lv.w);
  }
  __syncthreads();
  if (tid < NEDGE) {
    if (lcnt[tid]) atomicAdd(&binCnt[c * NEDGE + tid], lcnt[tid]);
    if (lsum[tid]) atomicAdd(&binLab[c * NEDGE + tid], lsum[tid]);
  }
}

// ---------------- K10: leave-one-out contribution + reduction ----------------
__global__ __launch_bounds__(256) void contrib_pass(
    const float* __restrict__ probsT, const int* __restrict__ labels,
    const float* __restrict__ edges, const unsigned* __restrict__ binLab,
    const unsigned* __restrict__ binCnt, float* __restrict__ out, int n) {
  __shared__ float wsum[8];
  __shared__ float bls[NBINS], bcs[NBINS];
  const int tid = threadIdx.x;
  const int c   = blockIdx.y;
  float e[NEDGE];
  for (int k = 0; k < NEDGE; ++k) e[k] = edges[c * NEDGE + k];
  if (tid < NBINS) {
    bls[tid] = (float)binLab[c * NEDGE + tid];
    bcs[tid] = (float)binCnt[c * NEDGE + tid];
  }
  __syncthreads();
  const size_t base = (size_t)c * n + (size_t)blockIdx.x * CHUNK;
  const float4* p4 = (const float4*)(probsT + base);
  const int4*   l4 = (const int4*)(labels + (size_t)blockIdx.x * CHUNK);
  float acc = 0.0f;
  auto doone = [&](float conf, int lab) {
    int idx = 0;
#pragma unroll
    for (int k = 0; k < NEDGE; ++k) idx += (e[k] < conf);
    if (idx >= 1) {
      const int bin = idx - 1;
      const float lf = (lab == c) ? 1.0f : 0.0f;
      const float ce = (bls[bin] - lf) / (bcs[bin] - 1.0f);
      const float d  = conf - ce;
      acc += d * d;                    // P = 2
    }
  };
  for (int i = 0; i < CHUNK / 1024; ++i) {
    __builtin_prefetch(p4 + tid + 256 * (i + 2), 0, 1);
    const int vi = tid + 256 * i;
    const float4 v = p4[vi];
    const int4 lv = l4[vi];
    doone(v.x, lv.x); doone(v.y, lv.y); doone(v.z, lv.z); doone(v.w, lv.w);
  }
  for (int off = 16; off; off >>= 1) acc += __shfl_xor(acc, off, 32);
  if ((tid & 31) == 0) wsum[tid >> 5] = acc;
  __syncthreads();
  if (tid == 0) {
    float s = 0.0f;
    for (int w = 0; w < 8; ++w) s += wsum[w];
    atomicAdd(&out[c], s / (float)n);
  }
}

// ---------------- host launch ----------------
extern "C" void kernel_launch(void* const* d_in, const int* in_sizes, int n_in,
                              void* d_out, int out_size, void* d_ws, size_t ws_size,
                              hipStream_t stream) {
  const float* logits = (const float*)d_in[0];
  const int*   labels = (const int*)d_in[1];
  float*       out    = (float*)d_out;
  const int n = in_sizes[1];   // 262144 samples (labels count)

  char* ws = (char*)d_ws;
  const size_t oProbs = 0;
  const size_t oH1 = oProbs + (size_t)NCLS * (size_t)n * 4;
  const size_t oH2 = oH1 + (size_t)NCLS * H1BINS * 4;
  const size_t oH3 = oH2 + (size_t)NCLS * NTGT * H2BINS * 4;
  const size_t oBL = oH3 + (size_t)NCLS * NTGT * H3BINS * 4;
  const size_t oBC = oBL + (size_t)NCLS * NEDGE * 4;
  const size_t oZE = oBC + (size_t)NCLS * NEDGE * 4;   // end of zeroed region
  const size_t oB1 = oZE;
  const size_t oR1 = oB1 + (size_t)NCLS * NTGT * 4;
  const size_t oB2 = oR1 + (size_t)NCLS * NTGT * 4;
  const size_t oR2 = oB2 + (size_t)NCLS * NTGT * 4;
  const size_t oV  = oR2 + (size_t)NCLS * NTGT * 4;
  const size_t oE  = oV  + (size_t)NCLS * NTGT * 4;

  float*    probsT = (float*)(ws + oProbs);
  unsigned* h1     = (unsigned*)(ws + oH1);
  unsigned* h2     = (unsigned*)(ws + oH2);
  unsigned* h3     = (unsigned*)(ws + oH3);
  unsigned* bLab   = (unsigned*)(ws + oBL);
  unsigned* bCnt   = (unsigned*)(ws + oBC);
  unsigned* b1     = (unsigned*)(ws + oB1);
  unsigned* r1     = (unsigned*)(ws + oR1);
  unsigned* b2     = (unsigned*)(ws + oB2);
  unsigned* r2     = (unsigned*)(ws + oR2);
  float*    vals   = (float*)(ws + oV);
  float*    edges  = (float*)(ws + oE);

  const size_t zeroVec = (oZE - oH1) / 16;   // region size is 16B-divisible
  zero_ws<<<2048, 256, 0, stream>>>((uint4*)(ws + oH1), zeroVec, out, out_size);

  softmax_transpose<<<n / TILE_R, 256, 0, stream>>>(logits, probsT, n);

  dim3 g(n / CHUNK, NCLS);
  hist_l1<<<g, 256, 0, stream>>>(probsT, h1, n);
  select_l1<<<NCLS, 32, 0, stream>>>(h1, b1, r1, n);
  hist_l2<<<g, 256, 0, stream>>>(probsT, b1, h2, n);
  select_l2<<<NCLS, 32, 0, stream>>>(h2, r1, b2, r2);
  hist_l3<<<g, 256, 0, stream>>>(probsT, b1, b2, h3, n);
  select_l3<<<NCLS, 32, 0, stream>>>(h3, b1, b2, r2, vals);
  make_edges<<<(NCLS + 63) / 64, 64, 0, stream>>>(vals, edges, n);
  bin_stats<<<g, 256, 0, stream>>>(probsT, labels, edges, bLab, bCnt, n);
  contrib_pass<<<g, 256, 0, stream>>>(probsT, labels, edges, bLab, bCnt, out, n);
}
